// NonSpatialGNN_35407710389109
// MI455X (gfx1250) — compile-verified, matrix operand order checked by reading, not run on hardware
//
#include <hip/hip_runtime.h>
#include <hip/hip_bf16.h>

#define NN_ 10000
#define EE_ 40000   // must be a multiple of 16 (tile height of the edge GEMM)
#define DIN_ 11
#define DD_ 64
#define FE_ 4
#define LL_ 4
#define BB_ 512
#define MM_ 3

typedef __attribute__((ext_vector_type(16))) __bf16 v16bf;
typedef __attribute__((ext_vector_type(16))) unsigned short v16us;
typedef __attribute__((ext_vector_type(8))) float v8f;

__device__ __forceinline__ unsigned short f32_to_bf16(float f) {
    unsigned int u = __float_as_uint(f);
    unsigned int r = u + 0x7FFFu + ((u >> 16) & 1u);   // round-to-nearest-even
    return (unsigned short)(r >> 16);
}
__device__ __forceinline__ float bf16_lo_to_f32(unsigned int u) { return __uint_as_float(u << 16); }
__device__ __forceinline__ float bf16_hi_to_f32(unsigned int u) { return __uint_as_float(u & 0xFFFF0000u); }

// ---------------- pack Wn2 (fp32 [128,4096]) into WMMA B-fragment order, bf16 ----------------
// dst[((nt*4 + ks)*32 + lane)*16 + i] = bf16( Wn2[k*4096 + nt*16 + (lane&15)] )
//   with k = 32*ks + 16*(i>>3) + 8*(lane>>4) + (i&7)
// Each lane's 16 values become one contiguous 32-byte chunk -> global_load_b128 x2 in the GEMM.
__global__ void k_pack_b(const float* __restrict__ Wn2, unsigned short* __restrict__ dst) {
    int idx = blockIdx.x * blockDim.x + threadIdx.x;
    if (idx >= 128 * 4096) return;
    int i    = idx & 15;
    int lane = (idx >> 4) & 31;
    int ks   = (idx >> 9) & 3;
    int nt   = idx >> 11;
    int k    = ks * 32 + ((i >> 3) << 4) + ((lane >> 4) << 3) + (i & 7);
    int col  = nt * 16 + (lane & 15);
    dst[idx] = f32_to_bf16(Wn2[k * 4096 + col]);
}

// ---------------- h0 = x @ W0 + b0  [N,11]@[11,64] ----------------
__global__ void k_h0(const float* __restrict__ x, const float* __restrict__ W0,
                     const float* __restrict__ b0, float* __restrict__ h) {
    int t = blockIdx.x * blockDim.x + threadIdx.x;
    if (t >= NN_ * DD_) return;
    int n = t >> 6, d = t & 63;
    float v = b0[d];
#pragma unroll
    for (int k = 0; k < DIN_; ++k) v += x[n * DIN_ + k] * W0[k * DD_ + d];
    h[t] = v;
}

// ---------------- edge MLP + big GEMM via WMMA ----------------
// ew[e, :, :] = silu(edge_attr[e]@Wn1 + bn1) @ Wn2 + bn2   (stored bf16, row-major [E][4096])
// E is a multiple of 16, so the 16-edge tile never straddles the end: no guards needed.
__global__ void k_edge_gemm(const float* __restrict__ ea, const float* __restrict__ Wn1,
                            const float* __restrict__ bn1, const unsigned short* __restrict__ wn2pk,
                            const float* __restrict__ bn2, unsigned short* __restrict__ ew) {
    // A tile in WMMA fragment order: sA[(ks*32 + lane)*16 + i]
    __shared__ alignas(32) unsigned short sA[4 * 32 * 16];
    const int e0 = blockIdx.x * 16;
    const int tid = threadIdx.x;

    // Phase 1: hidden activations for 16 edges (tiny GEMM + silu), scattered into fragment order
    for (int idx = tid; idx < 16 * 128; idx += 128) {
        int el = idx >> 7, c = idx & 127;     // el = row (edge in tile), c = K index
        int e = e0 + el;
        float v = bn1[c];
#pragma unroll
        for (int k = 0; k < FE_; ++k) v += ea[e * FE_ + k] * Wn1[k * 128 + c];
        v = v / (1.f + __expf(-v));  // silu
        int ks = c >> 5, r5 = c & 31;
        int fi = ((r5 >> 4) & 1) * 8 + (r5 & 7);
        int half = (r5 >> 3) & 1;
        int lane_s = half * 16 + el;
        sA[((ks * 32 + lane_s) << 4) + fi] = f32_to_bf16(v);
    }
    __syncthreads();

    const int wave = tid >> 5, lane = tid & 31;
    const int half = lane >> 4, nn = lane & 15;

    // A fragments: two ds_load_b128 per K-step, loaded once, reused for all 256 N-tiles.
    v16bf aF[4];
#pragma unroll
    for (int ks = 0; ks < 4; ++ks) {
        v16us au = *(const v16us*)&sA[(ks * 32 + lane) << 4];
        aF[ks] = __builtin_bit_cast(v16bf, au);
    }

    for (int nt = wave; nt < 256; nt += 4) {
        const int c0 = nt * 16;
        float bias = bn2[c0 + nn];
        v8f acc;
#pragma unroll
        for (int r = 0; r < 8; ++r) acc[r] = bias;
#pragma unroll
        for (int ks = 0; ks < 4; ++ks) {
            // B fragment: one contiguous 32-byte chunk per lane (coalesced across lanes)
            v16us bu = *(const v16us*)&wn2pk[((nt * 4 + ks) * 32 + lane) << 4];
            v16bf bF = __builtin_bit_cast(v16bf, bu);
            acc = __builtin_amdgcn_wmma_f32_16x16x32_bf16(false, aF[ks], false, bF,
                                                          (short)0, acc, false, false);
        }
        // D layout: M = vgpr + 8*(lane>>4), N = lane&15. Unguarded straight-line stores.
        unsigned short* __restrict__ dstp = ew + (size_t)(e0 + 8 * half) * 4096 + c0 + nn;
#pragma unroll
        for (int r = 0; r < 8; ++r) {
            dstp[(size_t)r * 4096] = f32_to_bf16(acc[r]);
        }
    }
}

// ---------------- per-edge mat-vec + scatter-add (memory bound) ----------------
__global__ void k_msg(const float* __restrict__ h, const unsigned short* __restrict__ ew,
                      const int* __restrict__ ei, float* __restrict__ aggr) {
    int wid = (blockIdx.x * blockDim.x + threadIdx.x) >> 5;
    int lane = threadIdx.x & 31;
    if (wid >= EE_) return;
    int src = ei[wid];
    int dst = ei[EE_ + wid];
    float hlo = h[src * DD_ + lane];
    float hhi = h[src * DD_ + 32 + lane];
    const unsigned int* ewr = (const unsigned int*)(ew + (size_t)wid * 4096);
    float a0 = 0.f, a1 = 0.f;
#pragma unroll 4
    for (int k = 0; k < 32; ++k) {
        float hk = __shfl(hlo, k, 32);
        unsigned int u = ewr[k * 32 + lane];  // cols 2*lane, 2*lane+1 of row k
        a0 += hk * bf16_lo_to_f32(u);
        a1 += hk * bf16_hi_to_f32(u);
    }
#pragma unroll 4
    for (int k = 0; k < 32; ++k) {
        float hk = __shfl(hhi, k, 32);
        unsigned int u = ewr[(32 + k) * 32 + lane];
        a0 += hk * bf16_lo_to_f32(u);
        a1 += hk * bf16_hi_to_f32(u);
    }
    atomicAdd(&aggr[(size_t)dst * DD_ + 2 * lane], a0);
    atomicAdd(&aggr[(size_t)dst * DD_ + 2 * lane + 1], a1);
}

// ---------------- h = h + aggr + h@root_l + bias_l ----------------
__global__ void k_update(float* __restrict__ h, const float* __restrict__ aggr,
                         const float* __restrict__ root_l, const float* __restrict__ bias_l) {
    int wid = (blockIdx.x * blockDim.x + threadIdx.x) >> 5;
    int lane = threadIdx.x & 31;
    if (wid >= NN_) return;
    float* hr = h + (size_t)wid * DD_;
    float hlo = hr[lane], hhi = hr[32 + lane];
    float a0 = 0.f, a1 = 0.f;
#pragma unroll 4
    for (int k = 0; k < 32; ++k) {
        float hk = __shfl(hlo, k, 32);
        const float* rr = root_l + k * DD_;
        a0 += hk * rr[2 * lane];
        a1 += hk * rr[2 * lane + 1];
    }
#pragma unroll 4
    for (int k = 0; k < 32; ++k) {
        float hk = __shfl(hhi, k, 32);
        const float* rr = root_l + (32 + k) * DD_;
        a0 += hk * rr[2 * lane];
        a1 += hk * rr[2 * lane + 1];
    }
    // old h at cols 2*lane, 2*lane+1 via shuffles (avoid reload/alias with in-place store)
    int c0 = 2 * lane, c1 = 2 * lane + 1;
    float v0a = __shfl(hlo, c0 & 31, 32), v0b = __shfl(hhi, c0 & 31, 32);
    float v1a = __shfl(hlo, c1 & 31, 32), v1b = __shfl(hhi, c1 & 31, 32);
    float hc0 = (lane < 16) ? v0a : v0b;
    float hc1 = (lane < 16) ? v1a : v1b;
    hr[c0] = hc0 + aggr[(size_t)wid * DD_ + c0] + a0 + bias_l[c0];
    hr[c1] = hc1 + aggr[(size_t)wid * DD_ + c1] + a1 + bias_l[c1];
}

// ---------------- Set2Set + readout: one block per graph ----------------
__device__ __forceinline__ int lower_bound_i(const int* a, int n, int v) {
    int lo = 0, hi = n;
    while (lo < hi) { int mid = (lo + hi) >> 1; if (a[mid] < v) lo = mid + 1; else hi = mid; }
    return lo;
}

__global__ void __launch_bounds__(64)
k_set2set(const float* __restrict__ h, const int* __restrict__ batch,
          const float* __restrict__ Wih, const float* __restrict__ Whh,
          const float* __restrict__ lb, const float* __restrict__ Wo1,
          const float* __restrict__ bo1, const float* __restrict__ Wo2,
          const float* __restrict__ bo2, float* __restrict__ out) {
    __shared__ float sQ[2 * DD_];   // q_star
    __shared__ float sHs[DD_], sCs[DD_], sG[4 * DD_];
    __shared__ float sE[64], sRed[64];
    const int b = blockIdx.x, t = threadIdx.x;

    const int start = lower_bound_i(batch, NN_, b);
    const int end   = lower_bound_i(batch, NN_, b + 1);

    sHs[t] = 0.f; sCs[t] = 0.f; sQ[t] = 0.f; sQ[DD_ + t] = 0.f;
    __syncthreads();

    for (int m = 0; m < MM_; ++m) {
        // LSTM gates: gates = q_star@Wih^T + hs@Whh^T + b
        for (int j = t; j < 4 * DD_; j += 64) {
            float g = lb[j];
            for (int k = 0; k < 2 * DD_; ++k) g += sQ[k] * Wih[j * (2 * DD_) + k];
            for (int k = 0; k < DD_; ++k) g += sHs[k] * Whh[j * DD_ + k];
            sG[j] = g;
        }
        __syncthreads();
        {
            float ig = sG[t], fg = sG[DD_ + t], gg = sG[2 * DD_ + t], og = sG[3 * DD_ + t];
            float si = 1.f / (1.f + __expf(-ig)), sf = 1.f / (1.f + __expf(-fg));
            float so = 1.f / (1.f + __expf(-og));
            float cs = sf * sCs[t] + si * tanhf(gg);
            sCs[t] = cs;
            sHs[t] = so * tanhf(cs);
        }
        __syncthreads();

        // online-softmax attention over this graph's nodes (chunks of 64)
        float runM = -INFINITY, runS = 0.f, racc = 0.f;  // thread t owns column d=t of r
        for (int base = start; base < end; base += 64) {
            int n = base + t;
            float e = -INFINITY;
            if (n < end) {
                e = 0.f;
                for (int d = 0; d < DD_; ++d) e += h[(size_t)n * DD_ + d] * sHs[d];
            }
            // chunk max
            sRed[t] = e; __syncthreads();
            for (int o = 32; o >= 1; o >>= 1) { if (t < o) sRed[t] = fmaxf(sRed[t], sRed[t + o]); __syncthreads(); }
            float cmax = sRed[0]; __syncthreads();
            float newM = fmaxf(runM, cmax);
            float scale = __expf(runM - newM);   // 0 when runM==-inf
            float ex = (n < end) ? __expf(e - newM) : 0.f;
            sE[t] = ex;
            // chunk sum
            sRed[t] = ex; __syncthreads();
            for (int o = 32; o >= 1; o >>= 1) { if (t < o) sRed[t] += sRed[t + o]; __syncthreads(); }
            float csum = sRed[0]; __syncthreads();
            runS = runS * scale + csum;
            runM = newM;
            // r accumulation: thread t = column d
            int cnt = min(64, end - base);
            float part = 0.f;
            for (int j = 0; j < cnt; ++j) part += sE[j] * h[(size_t)(base + j) * DD_ + t];
            racc = racc * scale + part;
            __syncthreads();  // protect sE before next chunk
        }
        float r = (runS > 0.f) ? (racc / runS) : 0.f;
        sQ[t] = sHs[t];
        sQ[DD_ + t] = r;
        __syncthreads();
    }

    // readout: silu(q_star@Wo1 + bo1) @ Wo2 + bo2
    float hid = bo1[t];
    for (int k = 0; k < 2 * DD_; ++k) hid += sQ[k] * Wo1[k * DD_ + t];
    hid = hid / (1.f + __expf(-hid));
    float part = hid * Wo2[t];
    sRed[t] = part; __syncthreads();
    for (int o = 32; o >= 1; o >>= 1) { if (t < o) sRed[t] += sRed[t + o]; __syncthreads(); }
    if (t == 0) out[b] = sRed[0] + bo2[0];
}

extern "C" void kernel_launch(void* const* d_in, const int* in_sizes, int n_in,
                              void* d_out, int out_size, void* d_ws, size_t ws_size,
                              hipStream_t stream) {
    const float* x         = (const float*)d_in[0];
    const int*   edge_index= (const int*)d_in[1];
    const float* edge_attr = (const float*)d_in[2];
    const int*   batch     = (const int*)d_in[3];
    const float* W0        = (const float*)d_in[4];
    const float* b0        = (const float*)d_in[5];
    const float* Wn1       = (const float*)d_in[6];
    const float* bn1       = (const float*)d_in[7];
    const float* Wn2       = (const float*)d_in[8];
    const float* bn2       = (const float*)d_in[9];
    const float* root      = (const float*)d_in[10];
    const float* conv_bias = (const float*)d_in[11];
    const float* Wih       = (const float*)d_in[12];
    const float* Whh       = (const float*)d_in[13];
    const float* lstm_b    = (const float*)d_in[14];
    const float* Wo1       = (const float*)d_in[15];
    const float* bo1       = (const float*)d_in[16];
    const float* Wo2       = (const float*)d_in[17];
    const float* bo2       = (const float*)d_in[18];
    float* out = (float*)d_out;

    char* ws = (char*)d_ws;
    unsigned short* ew    = (unsigned short*)ws;                             // E*4096 bf16 = 327.7 MB
    unsigned short* wn2pk = (unsigned short*)(ws + (size_t)EE_ * 4096 * 2);  // 1 MB, fragment-packed
    float* h    = (float*)(ws + (size_t)EE_ * 4096 * 2 + (size_t)128 * 4096 * 2);
    float* aggr = h + (size_t)NN_ * DD_;

    k_pack_b<<<(128 * 4096 + 255) / 256, 256, 0, stream>>>(Wn2, wn2pk);
    k_h0<<<(NN_ * DD_ + 255) / 256, 256, 0, stream>>>(x, W0, b0, h);
    k_edge_gemm<<<EE_ / 16, 128, 0, stream>>>(edge_attr, Wn1, bn1, wn2pk, bn2, ew);

    for (int l = 0; l < LL_; ++l) {
        hipMemsetAsync(aggr, 0, (size_t)NN_ * DD_ * sizeof(float), stream);
        k_msg<<<(EE_ + 7) / 8, 256, 0, stream>>>(h, ew, edge_index, aggr);
        k_update<<<(NN_ + 7) / 8, 256, 0, stream>>>(h, aggr, root + l * DD_ * DD_, conv_bias + l * DD_);
    }

    k_set2set<<<BB_, 64, 0, stream>>>(h, batch, Wih, Whh, lstm_b, Wo1, bo1, Wo2, bo2, out);
}
